// GNNEncoder_62569083568894
// MI455X (gfx1250) — matrix operand order repaired and checked by reading, not compiled, not run on hardware
//
#include <hip/hip_runtime.h>
#include <hip/hip_bf16.h>
#include <math.h>

// GNN encoder block for MI455X (gfx1250, wave32, WMMA).
// All GEMMs run on v_wmma_f32_16x16x32_bf16 (fp32 accum); GELU/LN/residual fp32.

#define ZB   4
#define NN   1024
#define KNE  48
#define DVC  128
#define DMSG 384
#define XST  408   // LDS stride (ushorts) for 384-wide rows (16B aligned, conflict-free)
#define HST  136   // LDS stride (ushorts) for 128-wide rows
#define FST  520   // LDS stride (ushorts) for 512-wide rows
#define EPSF 1e-5f

typedef __attribute__((ext_vector_type(16))) __bf16 v16bf;
typedef __attribute__((ext_vector_type(8)))  float  v8f;

union BFrag { v16bf v; uint4 q[2]; };

__device__ __forceinline__ unsigned short f2bf(float f) {
  unsigned int u = __float_as_uint(f);
  u += 0x7FFFu + ((u >> 16) & 1u);           // round-to-nearest-even
  return (unsigned short)(u >> 16);
}
__device__ __forceinline__ float gelu_erf(float x) {
  return 0.5f * x * (1.0f + erff(x * 0.70710678118654752f));
}

// A fragment (16x32 bf16): lane holds row m = ln; K-chunks at k0+hf*8 and k0+16+hf*8.
__device__ __forceinline__ v16bf load_a_lds(const unsigned short* X, int stride,
                                            int row, int k0, int hf) {
  const unsigned short* p = X + row * stride + k0 + hf * 8;
  BFrag u;
  u.q[0] = *reinterpret_cast<const uint4*>(p);
  u.q[1] = *reinterpret_cast<const uint4*>(p + 16);
  return u.v;
}
// B fragment (32x16 bf16) from pre-transposed weights WT[N][K]: lane holds col n,
// K-chunks at k0+hf*8 and k0+16+hf*8.
__device__ __forceinline__ v16bf load_b_glb(const unsigned short* WT, int Kd,
                                            int col, int k0, int hf) {
  const unsigned short* p = WT + (size_t)col * Kd + k0 + hf * 8;
  BFrag u;
  u.q[0] = *reinterpret_cast<const uint4*>(p);
  u.q[1] = *reinterpret_cast<const uint4*>(p + 16);
  return u.v;
}
__device__ __forceinline__ v8f wmma_bf16(v16bf a, v16bf b, v8f c) {
  return __builtin_amdgcn_wmma_f32_16x16x32_bf16(false, a, false, b, (short)0, c,
                                                 false, false);
}
__device__ __forceinline__ float hred16(float s) {   // reduce 16 lanes of same half
#pragma unroll
  for (int m = 1; m < 16; m <<= 1) s += __shfl_xor(s, m, 32);
  return s;
}
__device__ __forceinline__ float wred32(float s) {
#pragma unroll
  for (int m = 16; m > 0; m >>= 1) s += __shfl_xor(s, m, 32);
  return s;
}

// ---- weight prep: fp32 [K][Ncol] -> bf16 transposed [Ncol][K] -----------------
__global__ void wprep(const float* __restrict__ src, unsigned short* __restrict__ dst,
                      int K, int Ncol) {
  int idx = blockIdx.x * blockDim.x + threadIdx.x;
  if (idx >= K * Ncol) return;
  int k = idx / Ncol, c = idx - k * Ncol;
  dst[(size_t)c * K + k] = f2bf(src[idx]);
}

// ---- node message pass: Mv = MLP3(Vi||Vj||E)*mask; V1 = LN(V + sum_k Mv) ------
__global__ __launch_bounds__(96) void gnn_node_msg(
    const float* __restrict__ V, const float* __restrict__ E,
    const float* __restrict__ emask, const int* __restrict__ Kidx,
    const unsigned short* __restrict__ w1T, const float* __restrict__ b1,
    const unsigned short* __restrict__ w2T, const float* __restrict__ b2,
    const unsigned short* __restrict__ w3T, const float* __restrict__ b3,
    const float* __restrict__ g, const float* __restrict__ bta,
    float* __restrict__ Vout) {
  __shared__ __align__(16) unsigned short Xs[KNE * XST];
  __shared__ __align__(16) unsigned short Hs[KNE * HST];
  __shared__ float vres[DVC];
  __shared__ float accs[DVC];
  __shared__ float fmsk[KNE];
  __shared__ int   kidx[KNE];

  const int tid = threadIdx.x;
  const int lane = tid & 31, wv = tid >> 5;
  const int hf = lane >> 4, ln = lane & 15;
  const int zn = blockIdx.x;          // z*N + n
  const int z = zn >> 10;

  const float* Vrow = V + (size_t)zn * DVC;
  for (int t = tid; t < DVC; t += 96) { vres[t] = Vrow[t]; accs[t] = 0.f; }
  for (int r = tid; r < KNE; r += 96) {
    fmsk[r] = emask[(size_t)zn * KNE + r];
    kidx[r] = Kidx[(size_t)zn * KNE + r];
  }
  __syncthreads();

  for (int r = 0; r < KNE; ++r) {      // pack X = [Vi || Vj || E] as bf16
    const float* Vj = V + ((size_t)(z * NN) + kidx[r]) * DVC;
    const float* Er = E + ((size_t)zn * KNE + r) * DVC;
    unsigned short* xr = &Xs[r * XST];
    for (int c = tid; c < DMSG; c += 96) {
      float f = (c < 128) ? vres[c] : ((c < 256) ? Vj[c - 128] : Er[c - 256]);
      xr[c] = f2bf(f);
    }
  }
  __syncthreads();

  const int r0 = wv * 16;
  // layer 1: [16x384]@[384x128] + GELU -> Hs
  {
    v8f acc[8];
#pragma unroll
    for (int t = 0; t < 8; ++t) acc[t] = (v8f){0, 0, 0, 0, 0, 0, 0, 0};
    for (int k0 = 0; k0 < DMSG; k0 += 32) {
      v16bf a = load_a_lds(Xs, XST, r0 + ln, k0, hf);
#pragma unroll
      for (int nt = 0; nt < 8; ++nt) {
        v16bf b = load_b_glb(w1T, DMSG, nt * 16 + ln, k0, hf);
        acc[nt] = wmma_bf16(a, b, acc[nt]);
      }
    }
#pragma unroll
    for (int nt = 0; nt < 8; ++nt) {
      const int col = nt * 16 + ln;
      const float bias = b1[col];
#pragma unroll
      for (int v = 0; v < 8; ++v) {
        const int row = r0 + v + hf * 8;
        Hs[row * HST + col] = f2bf(gelu_erf(acc[nt][v] + bias));
      }
    }
  }
  __syncthreads();
  // layer 2: [16x128]@[128x128] + GELU -> Xs (reused, stride HST)
  {
    v8f acc[8];
#pragma unroll
    for (int t = 0; t < 8; ++t) acc[t] = (v8f){0, 0, 0, 0, 0, 0, 0, 0};
    for (int k0 = 0; k0 < DVC; k0 += 32) {
      v16bf a = load_a_lds(Hs, HST, r0 + ln, k0, hf);
#pragma unroll
      for (int nt = 0; nt < 8; ++nt) {
        v16bf b = load_b_glb(w2T, DVC, nt * 16 + ln, k0, hf);
        acc[nt] = wmma_bf16(a, b, acc[nt]);
      }
    }
#pragma unroll
    for (int nt = 0; nt < 8; ++nt) {
      const int col = nt * 16 + ln;
      const float bias = b2[col];
#pragma unroll
      for (int v = 0; v < 8; ++v) {
        const int row = r0 + v + hf * 8;
        Xs[row * HST + col] = f2bf(gelu_erf(acc[nt][v] + bias));
      }
    }
  }
  __syncthreads();
  // layer 3: [16x128]@[128x128], mask, row-sum into accs
  {
    v8f acc[8];
#pragma unroll
    for (int t = 0; t < 8; ++t) acc[t] = (v8f){0, 0, 0, 0, 0, 0, 0, 0};
    for (int k0 = 0; k0 < DVC; k0 += 32) {
      v16bf a = load_a_lds(Xs, HST, r0 + ln, k0, hf);
#pragma unroll
      for (int nt = 0; nt < 8; ++nt) {
        v16bf b = load_b_glb(w3T, DVC, nt * 16 + ln, k0, hf);
        acc[nt] = wmma_bf16(a, b, acc[nt]);
      }
    }
#pragma unroll
    for (int nt = 0; nt < 8; ++nt) {
      const int col = nt * 16 + ln;
      const float bias = b3[col];
      float part = 0.f;
#pragma unroll
      for (int v = 0; v < 8; ++v) {
        const int row = r0 + v + hf * 8;
        part += (acc[nt][v] + bias) * fmsk[row];
      }
      atomicAdd(&accs[col], part);   // ds_add_f32
    }
  }
  __syncthreads();
  if (tid < 32) {                    // residual + LayerNorm for this node
    float y[4]; float s = 0.f;
#pragma unroll
    for (int j = 0; j < 4; ++j) {
      int col = tid * 4 + j;
      y[j] = vres[col] + accs[col];
      s += y[j];
    }
    s = wred32(s);
    float mean = s * (1.f / 128.f);
    float s2 = 0.f;
#pragma unroll
    for (int j = 0; j < 4; ++j) { float d = y[j] - mean; s2 += d * d; }
    s2 = wred32(s2);
    float rs = rsqrtf(s2 * (1.f / 128.f) + EPSF);
#pragma unroll
    for (int j = 0; j < 4; ++j) {
      int col = tid * 4 + j;
      Vout[(size_t)zn * DVC + col] = (y[j] - mean) * rs * g[col] + bta[col];
    }
  }
}

// ---- FFN: V2 = LN(V1 + W2·gelu(W1·V1 + b1) + b2), 16 rows per wave ------------
__global__ __launch_bounds__(32) void gnn_ffn(
    const float* __restrict__ Vin,
    const unsigned short* __restrict__ w1T, const float* __restrict__ b1,
    const unsigned short* __restrict__ w2T, const float* __restrict__ b2,
    const float* __restrict__ g, const float* __restrict__ bta,
    float* __restrict__ Vout) {
  __shared__ __align__(16) unsigned short Xb[16 * HST];
  __shared__ __align__(16) unsigned short Hb[16 * FST];
  __shared__ float resf[16 * DVC];

  const int lane = threadIdx.x;
  const int hf = lane >> 4, ln = lane & 15;
  const int row0 = blockIdx.x * 16;

  for (int idx = lane; idx < 16 * DVC; idx += 32) {
    int r = idx >> 7, c = idx & 127;
    float f = Vin[(size_t)(row0 + r) * DVC + c];
    resf[idx] = f;
    Xb[r * HST + c] = f2bf(f);
  }
  __syncthreads();
  // 128 -> 512 + GELU
  {
    v16bf af[4];
#pragma unroll
    for (int kk = 0; kk < 4; ++kk) af[kk] = load_a_lds(Xb, HST, ln, kk * 32, hf);
    for (int nt = 0; nt < 32; ++nt) {
      v8f acc = (v8f){0, 0, 0, 0, 0, 0, 0, 0};
#pragma unroll
      for (int kk = 0; kk < 4; ++kk) {
        v16bf b = load_b_glb(w1T, DVC, nt * 16 + ln, kk * 32, hf);
        acc = wmma_bf16(af[kk], b, acc);
      }
      const int col = nt * 16 + ln;
      const float bias = b1[col];
#pragma unroll
      for (int v = 0; v < 8; ++v) {
        const int row = v + hf * 8;
        Hb[row * FST + col] = f2bf(gelu_erf(acc[v] + bias));
      }
    }
  }
  __syncthreads();
  // 512 -> 128, residual + per-row LN straight from C fragments
  {
    v8f acc[8];
#pragma unroll
    for (int t = 0; t < 8; ++t) acc[t] = (v8f){0, 0, 0, 0, 0, 0, 0, 0};
    for (int k0 = 0; k0 < 512; k0 += 32) {
      v16bf a = load_a_lds(Hb, FST, ln, k0, hf);
#pragma unroll
      for (int nt = 0; nt < 8; ++nt) {
        v16bf b = load_b_glb(w2T, 512, nt * 16 + ln, k0, hf);
        acc[nt] = wmma_bf16(a, b, acc[nt]);
      }
    }
#pragma unroll
    for (int v = 0; v < 8; ++v) {
      const int row = v + hf * 8;
      float y[8]; float s = 0.f;
#pragma unroll
      for (int nt = 0; nt < 8; ++nt) {
        const int col = nt * 16 + ln;
        y[nt] = resf[row * DVC + col] + acc[nt][v] + b2[col];
        s += y[nt];
      }
      s = hred16(s);
      float mean = s * (1.f / 128.f);
      float s2 = 0.f;
#pragma unroll
      for (int nt = 0; nt < 8; ++nt) { float d = y[nt] - mean; s2 += d * d; }
      s2 = hred16(s2);
      float rs = rsqrtf(s2 * (1.f / 128.f) + EPSF);
#pragma unroll
      for (int nt = 0; nt < 8; ++nt) {
        const int col = nt * 16 + ln;
        Vout[(size_t)(row0 + row) * DVC + col] =
            (y[nt] - mean) * rs * g[col] + bta[col];
      }
    }
  }
}

// ---- edge message pass: E' = LN(E + MLP3(Vi||Vj||E)*mask) ---------------------
__global__ __launch_bounds__(96) void gnn_edge_msg(
    const float* __restrict__ V, const float* __restrict__ E,
    const float* __restrict__ emask, const int* __restrict__ Kidx,
    const unsigned short* __restrict__ w1T, const float* __restrict__ b1,
    const unsigned short* __restrict__ w2T, const float* __restrict__ b2,
    const unsigned short* __restrict__ w3T, const float* __restrict__ b3,
    const float* __restrict__ g, const float* __restrict__ bta,
    float* __restrict__ Eout) {
  __shared__ __align__(16) unsigned short Xs[KNE * XST];
  __shared__ __align__(16) unsigned short Hs[KNE * HST];
  __shared__ float vres[DVC];
  __shared__ float fmsk[KNE];
  __shared__ int   kidx[KNE];

  const int tid = threadIdx.x;
  const int lane = tid & 31, wv = tid >> 5;
  const int hf = lane >> 4, ln = lane & 15;
  const int zn = blockIdx.x;
  const int z = zn >> 10;

  const float* Vrow = V + (size_t)zn * DVC;
  for (int t = tid; t < DVC; t += 96) vres[t] = Vrow[t];
  for (int r = tid; r < KNE; r += 96) {
    fmsk[r] = emask[(size_t)zn * KNE + r];
    kidx[r] = Kidx[(size_t)zn * KNE + r];
  }
  __syncthreads();

  for (int r = 0; r < KNE; ++r) {
    const float* Vj = V + ((size_t)(z * NN) + kidx[r]) * DVC;
    const float* Er = E + ((size_t)zn * KNE + r) * DVC;
    unsigned short* xr = &Xs[r * XST];
    for (int c = tid; c < DMSG; c += 96) {
      float f = (c < 128) ? vres[c] : ((c < 256) ? Vj[c - 128] : Er[c - 256]);
      xr[c] = f2bf(f);
    }
  }
  __syncthreads();

  const int r0 = wv * 16;
  {
    v8f acc[8];
#pragma unroll
    for (int t = 0; t < 8; ++t) acc[t] = (v8f){0, 0, 0, 0, 0, 0, 0, 0};
    for (int k0 = 0; k0 < DMSG; k0 += 32) {
      v16bf a = load_a_lds(Xs, XST, r0 + ln, k0, hf);
#pragma unroll
      for (int nt = 0; nt < 8; ++nt) {
        v16bf b = load_b_glb(w1T, DMSG, nt * 16 + ln, k0, hf);
        acc[nt] = wmma_bf16(a, b, acc[nt]);
      }
    }
#pragma unroll
    for (int nt = 0; nt < 8; ++nt) {
      const int col = nt * 16 + ln;
      const float bias = b1[col];
#pragma unroll
      for (int v = 0; v < 8; ++v) {
        const int row = r0 + v + hf * 8;
        Hs[row * HST + col] = f2bf(gelu_erf(acc[nt][v] + bias));
      }
    }
  }
  __syncthreads();
  {
    v8f acc[8];
#pragma unroll
    for (int t = 0; t < 8; ++t) acc[t] = (v8f){0, 0, 0, 0, 0, 0, 0, 0};
    for (int k0 = 0; k0 < DVC; k0 += 32) {
      v16bf a = load_a_lds(Hs, HST, r0 + ln, k0, hf);
#pragma unroll
      for (int nt = 0; nt < 8; ++nt) {
        v16bf b = load_b_glb(w2T, DVC, nt * 16 + ln, k0, hf);
        acc[nt] = wmma_bf16(a, b, acc[nt]);
      }
    }
#pragma unroll
    for (int nt = 0; nt < 8; ++nt) {
      const int col = nt * 16 + ln;
      const float bias = b2[col];
#pragma unroll
      for (int v = 0; v < 8; ++v) {
        const int row = r0 + v + hf * 8;
        Xs[row * HST + col] = f2bf(gelu_erf(acc[nt][v] + bias));
      }
    }
  }
  __syncthreads();
  {
    v8f acc[8];
#pragma unroll
    for (int t = 0; t < 8; ++t) acc[t] = (v8f){0, 0, 0, 0, 0, 0, 0, 0};
    for (int k0 = 0; k0 < DVC; k0 += 32) {
      v16bf a = load_a_lds(Xs, HST, r0 + ln, k0, hf);
#pragma unroll
      for (int nt = 0; nt < 8; ++nt) {
        v16bf b = load_b_glb(w3T, DVC, nt * 16 + ln, k0, hf);
        acc[nt] = wmma_bf16(a, b, acc[nt]);
      }
    }
    // residual + per-row LN from registers
#pragma unroll
    for (int v = 0; v < 8; ++v) {
      const int row = r0 + v + hf * 8;
      const float mk = fmsk[row];
      const float* Er = E + ((size_t)zn * KNE + row) * DVC;
      float y[8]; float s = 0.f;
#pragma unroll
      for (int nt = 0; nt < 8; ++nt) {
        const int col = nt * 16 + ln;
        y[nt] = Er[col] + (acc[nt][v] + b3[col]) * mk;
        s += y[nt];
      }
      s = hred16(s);
      float mean = s * (1.f / 128.f);
      float s2 = 0.f;
#pragma unroll
      for (int nt = 0; nt < 8; ++nt) { float d = y[nt] - mean; s2 += d * d; }
      s2 = hred16(s2);
      float rs = rsqrtf(s2 * (1.f / 128.f) + EPSF);
#pragma unroll
      for (int nt = 0; nt < 8; ++nt) {
        const int col = nt * 16 + ln;
        Eout[((size_t)zn * KNE + row) * DVC + col] =
            (y[nt] - mean) * rs * g[col] + bta[col];
      }
    }
  }
}

extern "C" void kernel_launch(void* const* d_in, const int* in_sizes, int n_in,
                              void* d_out, int out_size, void* d_ws, size_t ws_size,
                              hipStream_t stream) {
  const float* V        = (const float*)d_in[0];
  const float* E        = (const float*)d_in[1];
  const float* emask    = (const float*)d_in[2];
  const float* nm_w1    = (const float*)d_in[3];
  const float* nm_b1    = (const float*)d_in[4];
  const float* nm_w2    = (const float*)d_in[5];
  const float* nm_b2    = (const float*)d_in[6];
  const float* nm_w3    = (const float*)d_in[7];
  const float* nm_b3    = (const float*)d_in[8];
  const float* nmn_g    = (const float*)d_in[9];
  const float* nmn_b    = (const float*)d_in[10];
  const float* ffn_w1   = (const float*)d_in[11];
  const float* ffn_b1   = (const float*)d_in[12];
  const float* ffn_w2   = (const float*)d_in[13];
  const float* ffn_b2   = (const float*)d_in[14];
  const float* fn_g     = (const float*)d_in[15];
  const float* fn_b     = (const float*)d_in[16];
  const float* em_w1    = (const float*)d_in[17];
  const float* em_b1    = (const float*)d_in[18];
  const float* em_w2    = (const float*)d_in[19];
  const float* em_b2    = (const float*)d_in[20];
  const float* em_w3    = (const float*)d_in[21];
  const float* em_b3    = (const float*)d_in[22];
  const float* emn_g    = (const float*)d_in[23];
  const float* emn_b    = (const float*)d_in[24];
  const int*   Kidx     = (const int*)d_in[25];

  // workspace layout: bf16-transposed weights, then V1 (fp32)
  unsigned short* ws   = (unsigned short*)d_ws;
  unsigned short* nm1T = ws;                     // [128][384]
  unsigned short* nm2T = nm1T + 128 * 384;       // [128][128]
  unsigned short* nm3T = nm2T + 128 * 128;
  unsigned short* em1T = nm3T + 128 * 128;       // [128][384]
  unsigned short* em2T = em1T + 128 * 384;
  unsigned short* em3T = em2T + 128 * 128;
  unsigned short* f1T  = em3T + 128 * 128;       // [512][128]
  unsigned short* f2T  = f1T + 512 * 128;        // [128][512]
  float*          V1   = (float*)(f2T + 128 * 512);   // [4096][128]

  float* Vout = (float*)d_out;                   // [Z*N][128]
  float* Eout = Vout + (size_t)ZB * NN * DVC;    // [Z*N*Kn][128]

  wprep<<<(384 * 128 + 255) / 256, 256, 0, stream>>>(nm_w1, nm1T, 384, 128);
  wprep<<<(128 * 128 + 255) / 256, 256, 0, stream>>>(nm_w2, nm2T, 128, 128);
  wprep<<<(128 * 128 + 255) / 256, 256, 0, stream>>>(nm_w3, nm3T, 128, 128);
  wprep<<<(384 * 128 + 255) / 256, 256, 0, stream>>>(em_w1, em1T, 384, 128);
  wprep<<<(128 * 128 + 255) / 256, 256, 0, stream>>>(em_w2, em2T, 128, 128);
  wprep<<<(128 * 128 + 255) / 256, 256, 0, stream>>>(em_w3, em3T, 128, 128);
  wprep<<<(128 * 512 + 255) / 256, 256, 0, stream>>>(ffn_w1, f1T, 128, 512);
  wprep<<<(512 * 128 + 255) / 256, 256, 0, stream>>>(ffn_w2, f2T, 512, 128);

  gnn_node_msg<<<ZB * NN, 96, 0, stream>>>(V, E, emask, Kidx,
                                           nm1T, nm_b1, nm2T, nm_b2, nm3T, nm_b3,
                                           nmn_g, nmn_b, V1);
  gnn_ffn<<<(ZB * NN) / 16, 32, 0, stream>>>(V1, f1T, ffn_b1, f2T, ffn_b2,
                                             fn_g, fn_b, Vout);
  gnn_edge_msg<<<ZB * NN, 96, 0, stream>>>(Vout, E, emask, Kidx,
                                           em1T, em_b1, em2T, em_b2, em3T, em_b3,
                                           emn_g, emn_b, Eout);
}
